// DifferenceEqLayer_23742579212959
// MI455X (gfx1250) — matrix operand order repaired and checked by reading, not compiled, not run on hardware
//
#include <hip/hip_runtime.h>
#include <hip/hip_bf16.h>

// DifferenceEqLayer scan for MI455X (gfx1250, wave32, WMMA).
// T=1024, B=256, D=32, H=128.
//
// Grid: 16 workgroups (one per 16-row B tile), 256 threads = 8 waves.
// Wave w owns output columns [16w, 16w+16) of H. Running state h lives in
// f32 C-layout registers; each step it is mirrored to LDS as bf16 (ping-pong
// double buffer) so all waves can rebuild WMMA A-fragments (K = 128).
// W_h -> 4 resident B-fragments, W_in -> 1 resident B-fragment (VGPRs).
// Per step per wave: 10 ds_load_b128 (claused), ONE s_wait_dscnt, then
// 5 x v_wmma_f32_16x16x32_bf16 back-to-back in two accumulator chains;
// ONE dscnt-only split barrier; software-pipelined global load of next x.

#define T_DIM 1024
#define B_DIM 256
#define D_DIM 32
#define H_DIM 128
#define HSTR  136   // LDS h row stride in bf16 (128+8 -> conflict-free b128 reads)
#define XSTR  40    // LDS x row stride in bf16 (32+8)

typedef __bf16 bf16;
typedef __attribute__((ext_vector_type(16))) __bf16 v16bf;
typedef __attribute__((ext_vector_type(8)))  __bf16 v8bf;
typedef __attribute__((ext_vector_type(2)))  __bf16 v2bf;
typedef __attribute__((ext_vector_type(8)))  float  v8f;
typedef __attribute__((ext_vector_type(2)))  float  v2f;

__device__ __forceinline__ float fast_tanh(float x) {
#if __has_builtin(__builtin_amdgcn_tanhf)
  return __builtin_amdgcn_tanhf(x);        // v_tanh_f32 on gfx1250
#elif __has_builtin(__builtin_amdgcn_tanh_f32)
  return __builtin_amdgcn_tanh_f32(x);
#else
  return tanhf(x);
#endif
}

// Lightweight workgroup barrier: orders LDS only (s_wait_dscnt 0), does NOT
// drain global storecnt like __syncthreads' full fence would. The NT stores
// of `out` never need to be ordered against the barrier.
__device__ __forceinline__ void wg_sync_lds() {
#if __has_builtin(__builtin_amdgcn_s_barrier_signal) && \
    __has_builtin(__builtin_amdgcn_s_barrier_wait)
  asm volatile("s_wait_dscnt 0" ::: "memory");
  __builtin_amdgcn_s_barrier_signal(-1);
  __builtin_amdgcn_s_barrier_wait((short)-1);
  asm volatile("" ::: "memory");
#else
  __syncthreads();
#endif
}

// Build a 16x32 bf16 A-fragment from an LDS tile (row-major, stride `stride`
// bf16), K window starting at kbase. Layout: lanes 0-15 hold K=kbase+0..15;
// lanes 16-31 hold K=kbase+16..31 (matches B-fragment order).
__device__ __forceinline__ v16bf load_a_frag(const bf16* lds, int stride,
                                             int m, int half, int kbase) {
  const v8bf lo = *(const v8bf*)(lds + m * stride + kbase + 16 * half);
  const v8bf hi = *(const v8bf*)(lds + m * stride + kbase + 16 * half + 8);
  return __builtin_shufflevector(lo, hi, 0, 1, 2, 3, 4, 5, 6, 7,
                                 8, 9, 10, 11, 12, 13, 14, 15);
}

__global__ __launch_bounds__(256)
void diffeq_scan_kernel(const float* __restrict__ x,
                        const float* __restrict__ state,
                        const float* __restrict__ W_in,
                        const float* __restrict__ W_h,
                        const float* __restrict__ bias,
                        float* __restrict__ out,      // [T, B, H]
                        float* __restrict__ finals) { // [B, H]
  // Ping-pong buffers: step t reads [t&1], writes [(t+1)&1] -> one barrier/step.
  __shared__ __align__(16) bf16 hT[2][16 * HSTR];
  __shared__ __align__(16) bf16 xT[2][16 * XSTR];

  const int tid   = threadIdx.x;
  const int wave  = tid >> 5;      // 0..7 -> N tile
  const int lane  = tid & 31;
  const int nloc  = lane & 15;
  const int half  = lane >> 4;
  const int col   = wave * 16 + nloc;         // H column owned by this lane
  const int brow0 = blockIdx.x * 16;          // B tile base row
  const int m     = lane & 15;                // A-frag row

  // x staging geometry: 256 threads x 2 contiguous floats = 16x32 tile.
  const int xrow = (tid * 2) >> 5;            // 0..15
  const int xd   = (tid * 2) & 31;            // 0..31, even

  // ---- Preload resident B-fragments for W_h (K=0..127 -> 4 frags) and
  // W_in (K=0..31 -> 1 frag). B layout used here (contiguous): element e -> K = kbase+16*half+e,
  // lane nloc -> N = col. One-time scalar gathers, converted to bf16.
  v16bf bh[4];
#pragma unroll
  for (int f = 0; f < 4; ++f) {
#pragma unroll
    for (int e = 0; e < 16; ++e) {
      const int k = 32 * f + 16 * half + e;
      bh[f][e] = (bf16)W_h[k * H_DIM + col];
    }
  }
  v16bf bin;
#pragma unroll
  for (int e = 0; e < 16; ++e) {
    const int k = 16 * half + e;
    bin[e] = (bf16)W_in[k * H_DIM + col];
  }
  const float bval = bias[col];

  // ---- Initial state: f32 in C-layout registers (elem r -> M = r + 8*half),
  // mirrored to LDS buffer 0 as bf16.
  float hreg[8];
#pragma unroll
  for (int r = 0; r < 8; ++r) {
    const int mrow = r + 8 * half;
    hreg[r] = state[(size_t)(brow0 + mrow) * H_DIM + col];
    hT[0][mrow * HSTR + col] = (bf16)hreg[r];
  }
  {
    // Stage x_0 into xT[0].
    const v2f v = __builtin_nontemporal_load(
        (const v2f*)(x + (((size_t)0 * B_DIM + brow0 + xrow) * D_DIM + xd)));
    v2bf p; p[0] = (bf16)v[0]; p[1] = (bf16)v[1];
    *(v2bf*)(&xT[0][xrow * XSTR + xd]) = p;
  }
  __syncthreads();  // once, outside the hot loop

#pragma unroll 2    // makes cur/nxt compile-time constants
  for (int t = 0; t < T_DIM; ++t) {
    const int cur = t & 1;
    const int nxt = cur ^ 1;
    const bool havenext = (t + 1 < T_DIM);

    // Software pipeline: issue the raw f32 load of x_{t+1} NOW; its wait is
    // deferred to the convert at the tail, hidden under WMMA/tanh/stores.
    v2f xnext = {0.f, 0.f};
    if (havenext) {
      xnext = __builtin_nontemporal_load(
          (const v2f*)(x +
                       (((size_t)(t + 1) * B_DIM + brow0 + xrow) * D_DIM + xd)));
    }

    // Rebuild A-fragments from the current LDS buffers.
    v16bf xa  = load_a_frag(&xT[cur][0], XSTR, m, half, 0);
    v16bf ha0 = load_a_frag(&hT[cur][0], HSTR, m, half, 0);
    v16bf ha1 = load_a_frag(&hT[cur][0], HSTR, m, half, 32);
    v16bf ha2 = load_a_frag(&hT[cur][0], HSTR, m, half, 64);
    v16bf ha3 = load_a_frag(&hT[cur][0], HSTR, m, half, 96);
    // Pin all five fragments live here: forces all 10 ds_load_b128 to issue
    // up front with a single s_wait_dscnt, so the 5 WMMAs run back-to-back
    // instead of load->wait->wmma serialization.
    asm volatile("" : "+v"(xa), "+v"(ha0), "+v"(ha1), "+v"(ha2), "+v"(ha3));

    // Two independent accumulator chains (merged with 4 v_pk_add_f32).
    v8f acc0 = {bval, bval, bval, bval, bval, bval, bval, bval};
    v8f acc1 = {0.f, 0.f, 0.f, 0.f, 0.f, 0.f, 0.f, 0.f};
    acc0 = __builtin_amdgcn_wmma_f32_16x16x32_bf16(
        false, xa, false, bin, (short)0, acc0, false, false);
    acc1 = __builtin_amdgcn_wmma_f32_16x16x32_bf16(
        false, ha1, false, bh[1], (short)0, acc1, false, false);
    acc0 = __builtin_amdgcn_wmma_f32_16x16x32_bf16(
        false, ha0, false, bh[0], (short)0, acc0, false, false);
    acc1 = __builtin_amdgcn_wmma_f32_16x16x32_bf16(
        false, ha3, false, bh[3], (short)0, acc1, false, false);
    acc0 = __builtin_amdgcn_wmma_f32_16x16x32_bf16(
        false, ha2, false, bh[2], (short)0, acc0, false, false);
    const v8f acc = acc0 + acc1;  // 4 x v_pk_add_f32

    // Euler update + streamed output (non-temporal: write-once 134 MB).
    float nh[8];
#pragma unroll
    for (int r = 0; r < 8; ++r) {
      nh[r] = hreg[r] + fast_tanh(acc[r]);
      hreg[r] = nh[r];
      const int mrow = r + 8 * half;
      __builtin_nontemporal_store(
          nh[r], out + (((size_t)t * B_DIM + brow0 + mrow) * H_DIM + col));
    }

    // Publish h_{t+1} and x_{t+1} into the OTHER buffer (no WAR hazard:
    // that buffer's readers finished before the previous barrier).
#pragma unroll
    for (int r = 0; r < 8; ++r) {
      hT[nxt][(r + 8 * half) * HSTR + col] = (bf16)nh[r];
    }
    if (havenext) {
      v2bf p; p[0] = (bf16)xnext[0]; p[1] = (bf16)xnext[1];
      *(v2bf*)(&xT[nxt][xrow * XSTR + xd]) = p;
    }
    if (t + 8 < T_DIM) {
      __builtin_prefetch(
          x + (((size_t)(t + 8) * B_DIM + brow0 + xrow) * D_DIM + xd), 0, 1);
    }

    // One dscnt-guarded split barrier per step; global stores NOT waited on.
    wg_sync_lds();
  }

  // final_state output
#pragma unroll
  for (int r = 0; r < 8; ++r) {
    finals[(size_t)(brow0 + r + 8 * half) * H_DIM + col] = hreg[r];
  }
}

extern "C" void kernel_launch(void* const* d_in, const int* in_sizes, int n_in,
                              void* d_out, int out_size, void* d_ws, size_t ws_size,
                              hipStream_t stream) {
  (void)in_sizes; (void)n_in; (void)d_ws; (void)ws_size; (void)out_size;
  const float* x     = (const float*)d_in[0];
  const float* state = (const float*)d_in[1];
  const float* W_in  = (const float*)d_in[2];
  const float* W_h   = (const float*)d_in[3];
  const float* b     = (const float*)d_in[4];
  float* out    = (float*)d_out;                               // [T,B,H]
  float* finals = out + (size_t)T_DIM * B_DIM * H_DIM;         // [B,H]

  dim3 grid(B_DIM / 16);   // 16 independent B tiles
  dim3 block(256);         // 8 waves: one 16-col N tile each
  diffeq_scan_kernel<<<grid, block, 0, stream>>>(x, state, W_in, W_h, b, out,
                                                 finals);
}
